// SAAttention_6270652252164
// MI455X (gfx1250) — matrix operand order repaired and checked by reading, not compiled, not run on hardware
//
#include <hip/hip_runtime.h>
#include <hip/hip_bf16.h>
#include <math.h>

// ---------------------------------------------------------------------------
// Invariant Point Attention for MI455X (gfx1250, wave32, WMMA bf16 path)
// B=1, N=512, D=384, H=12, dk=32, P=128.
// out_pair reassociated: tmp[i,h,p] = sum_j attn[h,i,j]*x2d[i,j,p];
// out_pair = tmp @ w_pairv  (removes the 25.8 GFLOP v_pair materialization).
// Pair-bias GEMM accumulates transposed directly into attn logits (no pairb
// buffer, no strided softmax gather), staging x2d tiles through LDS with
// global_load_async_to_lds_b128 (builtin confirmed present by round-3 probe).
// ---------------------------------------------------------------------------

typedef __attribute__((ext_vector_type(16))) __bf16 v16bf;
typedef __attribute__((ext_vector_type(8)))  float  v8f;
typedef int v4i __attribute__((vector_size(16)));

#define NRES 512
#define NHEAD 12
#define DMODEL 384
#define DK 32
#define PDIM 128

#if defined(__has_builtin)
#  if __has_builtin(__builtin_amdgcn_global_load_async_to_lds_b128)
#    define HAVE_ASYNC_LDS 1
#  endif
#endif

// ---- A fragment: 16x32 bf16, K-contiguous. ISA 7.12.2 layout ---------------
// lanes 0-15: row=lane, elems 0-7 = K+0..7,  8-15 = K+16..23
// lanes16-31: row=lane-16, elems 0-7 = K+8..15, 8-15 = K+24..31
__device__ inline v16bf load_A_frag(const float* __restrict__ A, int row0, int k0,
                                    long sAr, int Mb, int lane) {
  const int half = lane >> 4;
  int r = row0 + (lane & 15);
  r = (r < Mb) ? r : (Mb - 1);                    // clamp only; no zeroing needed
  const float* p = A + (long)r * sAr + k0 + half * 8;
  float4 a0 = *(const float4*)(p + 0);
  float4 a1 = *(const float4*)(p + 4);
  float4 a2 = *(const float4*)(p + 16);
  float4 a3 = *(const float4*)(p + 20);
  v16bf a;
  a[0]=(__bf16)a0.x;  a[1]=(__bf16)a0.y;  a[2]=(__bf16)a0.z;  a[3]=(__bf16)a0.w;
  a[4]=(__bf16)a1.x;  a[5]=(__bf16)a1.y;  a[6]=(__bf16)a1.z;  a[7]=(__bf16)a1.w;
  a[8]=(__bf16)a2.x;  a[9]=(__bf16)a2.y;  a[10]=(__bf16)a2.z; a[11]=(__bf16)a2.w;
  a[12]=(__bf16)a3.x; a[13]=(__bf16)a3.y; a[14]=(__bf16)a3.z; a[15]=(__bf16)a3.w;
  return a;
}

// ---- B fragment: 32x16 bf16 ------------------------------------------------
// lanes 0-15: col=lane, elems 0-7 = K+0..7, 8-15 = K+16..23;  lanes 16-31: K+8..15, K+24..31
__device__ inline v16bf load_B_frag(const float* __restrict__ B, int k0, int col0,
                                    long sBr, long sBc, int Nb, int lane) {
  const int half = lane >> 4;
  int n = col0 + (lane & 15);
  n = (n < Nb) ? n : (Nb - 1);                    // clamp only
  const float* p = B + (long)n * sBc + (long)(k0 + half * 8) * sBr;
  v16bf b;
  if (sBr == 1) {                                 // K-contiguous: vector loads
    float4 b0 = *(const float4*)(p + 0);
    float4 b1 = *(const float4*)(p + 4);
    float4 b2 = *(const float4*)(p + 16);
    float4 b3 = *(const float4*)(p + 20);
    b[0]=(__bf16)b0.x;  b[1]=(__bf16)b0.y;  b[2]=(__bf16)b0.z;  b[3]=(__bf16)b0.w;
    b[4]=(__bf16)b1.x;  b[5]=(__bf16)b1.y;  b[6]=(__bf16)b1.z;  b[7]=(__bf16)b1.w;
    b[8]=(__bf16)b2.x;  b[9]=(__bf16)b2.y;  b[10]=(__bf16)b2.z; b[11]=(__bf16)b2.w;
    b[12]=(__bf16)b3.x; b[13]=(__bf16)b3.y; b[14]=(__bf16)b3.z; b[15]=(__bf16)b3.w;
  } else {                                        // strided K; coalesced over lanes
#pragma unroll
    for (int e = 0; e < 16; ++e) b[e] = (__bf16)p[(long)(e + (e & 8)) * sBr];
  }
  return b;
}

// ---- C/D store: 16x16 f32; VGPR r: lanes 0-15 -> (M=r,N=lane); 16-31 -> (M=8+r)
__device__ inline void store_C(float* __restrict__ C, v8f acc, int row0, int col0,
                               long ldc, int Mb, int Nb, float alpha,
                               const float* __restrict__ bias, int lane) {
  const int half = lane >> 4;
  const int n = col0 + (lane & 15);
  float bv = 0.0f;
  if (bias && n < Nb) bv = bias[n];
#pragma unroll
  for (int r = 0; r < 8; ++r) {
    int m = row0 + r + half * 8;
    if (m < Mb && n < Nb) C[(long)m * ldc + n] = alpha * acc[r] + bv;
  }
}

// ---- Strided batched GEMM: 1 wave computes a (16*MREP)x16 C tile -----------
template <int MREP>
__global__ void k_gemm(const float* __restrict__ A, const float* __restrict__ B,
                       float* __restrict__ C, int M, int N, int K,
                       long sAr, long sBr, long sBc, long ldc,
                       long Abatch, long Bbatch, long Cbatch,
                       float alpha, const float* __restrict__ bias) {
  const int lane = threadIdx.x;                    // blockDim.x == 32, EXEC full
  const int row0 = blockIdx.x * (16 * MREP);
  const int col0 = blockIdx.y * 16;
  const int b = blockIdx.z;
  const float* Ab = A + (long)b * Abatch;
  const float* Bb = B + (long)b * Bbatch;
  float* Cb = C + (long)b * Cbatch;

  v8f acc[MREP];
#pragma unroll
  for (int m = 0; m < MREP; ++m) acc[m] = (v8f){};

  for (int k0 = 0; k0 < K; k0 += 32) {             // K % 32 == 0 guaranteed
    if (k0 + 32 < K) {                             // -> global_prefetch_b8
      int r = row0 + (lane & 15);
      r = (r < M) ? r : (M - 1);
      __builtin_prefetch(&Ab[(long)r * sAr + (k0 + 32)], 0, 1);
    }
    v16bf bf = load_B_frag(Bb, k0, col0, sBr, sBc, N, lane);
#pragma unroll
    for (int m = 0; m < MREP; ++m) {
      v16bf af = load_A_frag(Ab, row0 + m * 16, k0, sAr, M, lane);
      acc[m] = __builtin_amdgcn_wmma_f32_16x16x32_bf16(false, af, false, bf,
                                                       (short)0, acc[m], false, false);
    }
  }
#pragma unroll
  for (int m = 0; m < MREP; ++m)
    store_C(Cb, acc[m], row0 + m * 16, col0, ldc, M, N, alpha, bias, lane);
}

// ---- Pair-bias GEMM, LDS-staged, transposed accumulate into attn -----------
// Per wave: 64 rows (ij) x K=128 of x2d staged to LDS (32 KB), 4 accumulators,
// result attn[h][ij] += pair_w * (x2d[ij,:] . w_pb[:,h]).
__global__ void k_gemm_pair(const float* __restrict__ x2d,
                            const float* __restrict__ w_pb,
                            float* __restrict__ attn, float pair_w) {
  const int lane = threadIdx.x;                    // 32 threads, EXEC full
  const int row0 = blockIdx.x * 64;
  alignas(16) __shared__ float tileA[64 * PDIM];   // 32 KB

  const float* gbase = x2d + (long)row0 * PDIM;
#if defined(HAVE_ASYNC_LDS)
  {
    typedef __attribute__((address_space(1))) v4i* gptr_t;
    typedef __attribute__((address_space(3))) v4i* lptr_t;
#pragma unroll 4
    for (int t = 0; t < 64; ++t) {
      int off = (t * 32 + lane) * 4;
      __builtin_amdgcn_global_load_async_to_lds_b128(
          (gptr_t)(gbase + off), (lptr_t)(tileA + off), 0, 0);
    }
  }
#  if __has_builtin(__builtin_amdgcn_s_wait_asynccnt)
  __builtin_amdgcn_s_wait_asynccnt(0);
#  else
  asm volatile("s_wait_asynccnt 0" ::: "memory");
#  endif
#else
#pragma unroll 4
  for (int t = 0; t < 64; ++t) {
    int off = (t * 32 + lane) * 4;
    *(float4*)(tileA + off) = *(const float4*)(gbase + off);
  }
#endif

  v8f acc[4];
#pragma unroll
  for (int m = 0; m < 4; ++m) acc[m] = (v8f){};

#pragma unroll
  for (int k0 = 0; k0 < PDIM; k0 += 32) {          // fully unrolled: 16 wmma
    v16bf bf = load_B_frag(w_pb, k0, 0, NHEAD, 1, NHEAD, lane);
#pragma unroll
    for (int m = 0; m < 4; ++m) {
      v16bf af = load_A_frag(tileA, m * 16, k0, PDIM, 64, lane);  // ds_load_b128
      acc[m] = __builtin_amdgcn_wmma_f32_16x16x32_bf16(false, af, false, bf,
                                                       (short)0, acc[m], false, false);
    }
  }

  // transposed accumulate: lane column n is head h; rows are consecutive ij
  const int half = lane >> 4;
  const int h = lane & 15;
  if (h < NHEAD) {
#pragma unroll
    for (int m = 0; m < 4; ++m) {
      float* dst = attn + (long)h * (NRES * NRES) + row0 + m * 16 + half * 8;
#pragma unroll
      for (int r = 0; r < 8; ++r) dst[r] += pair_w * acc[m][r];
    }
  }
}

// ---- Affine: glob[a] = sum_y pose_r[n,y,a]*l[y] + T[a] ----------------------
__global__ void k_affine(const float* __restrict__ loc, const float* __restrict__ Rm,
                         const float* __restrict__ T, float* __restrict__ glob, int npts) {
  int idx = blockIdx.x * blockDim.x + threadIdx.x;
  if (idx >= NRES * npts) return;
  int n = idx / npts, pt = idx % npts;
  const float* l = loc + ((long)n * npts + pt) * 3;
  const float* r = Rm + (long)n * 9;
  const float* t = T + (long)n * 3;
  float x = l[0], y = l[1], z = l[2];
  float* g = glob + ((long)n * npts + pt) * 3;
  g[0] = r[0] * x + r[3] * y + r[6] * z + t[0];
  g[1] = r[1] * x + r[4] * y + r[7] * z + t[1];
  g[2] = r[2] * x + r[5] * y + r[8] * z + t[2];
}

// ---- Fused logits (scalar+pair already in attn; add point + bias), softmax --
__global__ void k_logits_softmax(float* __restrict__ attn,
                                 const float* __restrict__ biasab,
                                 const float* __restrict__ qpg,
                                 const float* __restrict__ kpg,
                                 const float* __restrict__ tpw) {
  const int i = blockIdx.x, h = blockIdx.y, t = threadIdx.x;
  __shared__ float row[NRES];
  __shared__ float red[256];

  const float pw = 0.136082763f * log1pf(expf(tpw[h]));  // (1/sqrt(54))*softplus

  float qx[4], qy[4], qz[4];
  const float* qp = qpg + ((long)i * NHEAD + h) * 12;
#pragma unroll
  for (int p = 0; p < 4; ++p) { qx[p] = qp[p*3]; qy[p] = qp[p*3+1]; qz[p] = qp[p*3+2]; }

  float* arow = attn + ((long)h * NRES + i) * NRES;
  for (int j = t; j < NRES; j += 256) {
    const float* kp = kpg + ((long)j * NHEAD + h) * 12;
    float ds = 0.0f;
#pragma unroll
    for (int p = 0; p < 4; ++p) {
      float dx = qx[p] - kp[p*3], dy = qy[p] - kp[p*3+1], dz = qz[p] - kp[p*3+2];
      ds += sqrtf(dx*dx + dy*dy + dz*dz);
    }
    row[j] = arow[j] + biasab[(long)i * NRES + j] - 0.5f * pw * ds;
  }
  __syncthreads();

  float m = -1e30f;
  for (int j = t; j < NRES; j += 256) m = fmaxf(m, row[j]);
  red[t] = m; __syncthreads();
  for (int s = 128; s > 0; s >>= 1) { if (t < s) red[t] = fmaxf(red[t], red[t+s]); __syncthreads(); }
  m = red[0]; __syncthreads();

  float sum = 0.0f;
  for (int j = t; j < NRES; j += 256) { float e = __expf(row[j] - m); row[j] = e; sum += e; }
  red[t] = sum; __syncthreads();
  for (int s = 128; s > 0; s >>= 1) { if (t < s) red[t] += red[t+s]; __syncthreads(); }
  const float inv = 1.0f / red[0];
  for (int j = t; j < NRES; j += 256) arow[j] = row[j] * inv;
}

// ---- Inverse affine + norms -> feat[:,384:672) and feat[:,1056:1152) -------
__global__ void k_post(const float* __restrict__ opg, const float* __restrict__ Rm,
                       const float* __restrict__ T, float* __restrict__ feat) {
  int idx = blockIdx.x * blockDim.x + threadIdx.x;   // NRES * 96
  if (idx >= NRES * 96) return;
  int n = idx / 96, hp = idx % 96;
  const float* g = opg + (long)n * 288 + hp * 3;
  const float* r = Rm + (long)n * 9;
  const float* t = T + (long)n * 3;
  float dx = g[0]-t[0], dy = g[1]-t[1], dz = g[2]-t[2];
  float o0 = r[0]*dx + r[1]*dy + r[2]*dz;
  float o1 = r[3]*dx + r[4]*dy + r[5]*dz;
  float o2 = r[6]*dx + r[7]*dy + r[8]*dz;
  float* f = feat + (long)n * 1152;
  f[384 + hp*3 + 0] = o0;
  f[384 + hp*3 + 1] = o1;
  f[384 + hp*3 + 2] = o2;
  f[1056 + hp] = sqrtf(o0*o0 + o1*o1 + o2*o2);
}

// ---------------------------------------------------------------------------
static inline void gemm(hipStream_t s, const float* A, const float* B, float* C,
                        int M, int N, int K,
                        long sAr, long sBr, long sBc, long ldc,
                        long Ab, long Bb, long Cb, int batch,
                        float alpha, const float* bias) {
  if (M % 64 == 0) {
    dim3 g(M / 64, (N + 15) / 16, batch);
    hipLaunchKernelGGL((k_gemm<4>), g, dim3(32), 0, s,
                       A, B, C, M, N, K, sAr, sBr, sBc, ldc, Ab, Bb, Cb, alpha, bias);
  } else {
    dim3 g((M + 15) / 16, (N + 15) / 16, batch);
    hipLaunchKernelGGL((k_gemm<1>), g, dim3(32), 0, s,
                       A, B, C, M, N, K, sAr, sBr, sBc, ldc, Ab, Bb, Cb, alpha, bias);
  }
}

extern "C" void kernel_launch(void* const* d_in, const int* in_sizes, int n_in,
                              void* d_out, int out_size, void* d_ws, size_t ws_size,
                              hipStream_t stream) {
  (void)in_sizes; (void)n_in; (void)out_size; (void)ws_size;
  const float* x1d    = (const float*)d_in[0];
  const float* x2d    = (const float*)d_in[1];
  const float* pose_t = (const float*)d_in[2];
  const float* pose_r = (const float*)d_in[3];
  const float* biasab = (const float*)d_in[4];
  const float* w_sq   = (const float*)d_in[5];
  const float* w_sk   = (const float*)d_in[6];
  const float* w_sv   = (const float*)d_in[7];
  const float* w_pb   = (const float*)d_in[8];
  const float* w_pq   = (const float*)d_in[9];
  const float* w_pk   = (const float*)d_in[10];
  const float* w_pv   = (const float*)d_in[11];
  const float* tpw    = (const float*)d_in[12];
  const float* w_pairv= (const float*)d_in[13];
  const float* w_out  = (const float*)d_in[14];
  const float* b_out  = (const float*)d_in[15];
  float* out = (float*)d_out;

  float* ws = (float*)d_ws;
  float* qs   = ws + 0;          // 512*384
  float* ks_  = ws + 196608;     // 512*384
  float* vs   = ws + 393216;     // 512*384
  float* qpl  = ws + 589824;     // 512*144
  float* kpl  = ws + 663552;     // 512*144
  float* vpl  = ws + 737280;     // 512*288
  float* qpg  = ws + 884736;     // 512*144
  float* kpg  = ws + 958464;     // 512*144
  float* vpg  = ws + 1032192;    // 512*288
  float* attn = ws + 1179648;    // 12*512*512
  float* tmp  = ws + 8519680;    // 512*12*128
  float* feat = ws + 9306112;    // 512*1152
  float* opg  = ws + 9895936;    // 512*288

  const float scalar_w = 0.102062072616f;  // 1/sqrt(96)
  const float pair_w   = 0.5773502691896f; // 1/sqrt(3)

  // 1) scalar q/k/v projections: [512x384] @ [384x384]
  gemm(stream, x1d, w_sq, qs,  NRES, DMODEL, DMODEL, DMODEL, DMODEL,1, DMODEL, 0,0,0,1, 1.f, nullptr);
  gemm(stream, x1d, w_sk, ks_, NRES, DMODEL, DMODEL, DMODEL, DMODEL,1, DMODEL, 0,0,0,1, 1.f, nullptr);
  gemm(stream, x1d, w_sv, vs,  NRES, DMODEL, DMODEL, DMODEL, DMODEL,1, DMODEL, 0,0,0,1, 1.f, nullptr);

  // 2) point projections
  gemm(stream, x1d, w_pq, qpl, NRES, 144, DMODEL, DMODEL, 144,1, 144, 0,0,0,1, 1.f, nullptr);
  gemm(stream, x1d, w_pk, kpl, NRES, 144, DMODEL, DMODEL, 144,1, 144, 0,0,0,1, 1.f, nullptr);
  gemm(stream, x1d, w_pv, vpl, NRES, 288, DMODEL, DMODEL, 288,1, 288, 0,0,0,1, 1.f, nullptr);

  // 3) to global frame
  hipLaunchKernelGGL(k_affine, dim3((NRES*48+255)/256), dim3(256), 0, stream, qpl, pose_r, pose_t, qpg, 48);
  hipLaunchKernelGGL(k_affine, dim3((NRES*48+255)/256), dim3(256), 0, stream, kpl, pose_r, pose_t, kpg, 48);
  hipLaunchKernelGGL(k_affine, dim3((NRES*96+255)/256), dim3(256), 0, stream, vpl, pose_r, pose_t, vpg, 96);

  // 4) scalar attention logits: per head q_h @ k_h^T (B is K-contiguous)
  gemm(stream, qs, ks_, attn, NRES, NRES, DK,
       DMODEL, 1, DMODEL, NRES, DK, DK, (long)NRES*NRES, NHEAD, scalar_w, nullptr);

  // 5) pair bias: attn[h][ij] += pair_w * (x2d[ij,:] . w_pb[:,h])  (LDS-staged)
  hipLaunchKernelGGL(k_gemm_pair, dim3(NRES*NRES/64), dim3(32), 0, stream,
                     x2d, w_pb, attn, pair_w);

  // 6) fused point-attn + bias + softmax
  hipLaunchKernelGGL(k_logits_softmax, dim3(NRES, NHEAD), dim3(256), 0, stream,
                     attn, biasab, qpg, kpg, tpw);

  // 7) out_scalar: attn_h [512x512] @ v_s_h [512x32] -> feat[:, 0:384]
  gemm(stream, attn, vs, feat, NRES, DK, NRES,
       NRES, DMODEL, 1, 1152, (long)NRES*NRES, DK, DK, NHEAD, 1.f, nullptr);

  // 8) out_pg: attn_h @ v_p_h [512x24] -> opg [512x288] (global frame)
  gemm(stream, attn, vpg, opg, NRES, 24, NRES,
       NRES, 288, 1, 288, (long)NRES*NRES, 24, 24, NHEAD, 1.f, nullptr);

  // 9) tmp[i] = attn[:,i,:] [12x512] @ x2d[i] [512x128]  (batched over i)
  gemm(stream, attn, x2d, tmp, NHEAD, PDIM, NRES,
       (long)NRES*NRES, PDIM, 1, PDIM,
       NRES, (long)NRES*PDIM, (long)NHEAD*PDIM, NRES, 1.f, nullptr);

  // 10) out_pair: tmp_h [512x128] @ w_pairv_h [128x32] -> feat[:, 672:1056]
  gemm(stream, tmp, w_pairv, feat + 672, NRES, DK, PDIM,
       (long)NHEAD*PDIM, DMODEL, 1, 1152, PDIM, DK, DK, NHEAD, 1.f, nullptr);

  // 11) inverse affine + norms
  hipLaunchKernelGGL(k_post, dim3((NRES*96+255)/256), dim3(256), 0, stream,
                     opg, pose_r, pose_t, feat);

  // 12) output: feat [512x1152] @ w_out [1152x384] + b_out
  gemm(stream, feat, w_out, out, NRES, DMODEL, 1152,
       1152, DMODEL, 1, DMODEL, 0,0,0,1, 1.f, b_out);
}
